// NNUE_16930761081452
// MI455X (gfx1250) — matrix operand order repaired and checked by reading, not compiled, not run on hardware
//
#include <hip/hip_runtime.h>

// ---------------------------------------------------------------------------
// NNUE forward for MI455X (gfx1250, wave32, WMMA).
//   dominant op: [8192 x 49152] @ [49152 x 128] twice -> bf16 WMMA 16x16x32
//   HBM traffic ~1.61 GB (f32 features, each read once) -> ~69 us @ 23.3 TB/s
//   bf16 weights (25 MB) live in L2; staged into LDS per WG via
//   global_load_async_to_lds_b128 so the 8 waves share one copy. B-fragment
//   ds_loads are software-pipelined one 16x16 tile ahead of the WMMAs using
//   sched_group_barrier directives.
// ---------------------------------------------------------------------------

#define HALFK  24576
#define NUMELK 49152
#define BASEN  128
#define BATCH  8192
#define KTILES 768      // HALFK / 32
#define NTILES 8        // 128 / 16
#define NSTAGES 768     // total k-steps (2*KTILES) / 2 per stage

typedef __attribute__((ext_vector_type(16))) __bf16 v16bf;
typedef __attribute__((ext_vector_type(8)))  float  v8f;
typedef __attribute__((ext_vector_type(4)))  float  v4f;

// elements (bf16) per repacked weight matrix: kt * nt * 32 lanes * 16
static constexpr size_t FRAG_ELEMS = (size_t)KTILES * NTILES * 32 * 16; // 3,145,728
static constexpr size_t FRAG_BYTES = FRAG_ELEMS * 2;                    // 6,291,456

// Scheduling directives: 0x8 = MFMA/WMMA group, 0x100 = DS-read group.
#if __has_builtin(__builtin_amdgcn_sched_group_barrier)
#define SGB(MASK, SIZE, ID) __builtin_amdgcn_sched_group_barrier(MASK, SIZE, ID)
#else
#define SGB(MASK, SIZE, ID)
#endif

// ---------------------------------------------------------------------------
// Kernel 1: repack f32 weights -> bf16 WMMA-B fragments.
// B-matrix (32x16 bf16, wave32): lane L covers column n = nt*16 + (L&15),
// k-group g = L>>4 (e<8: k = kt*32 + g*8 + e; e>=8: k = kt*32 + 16 + g*8 + (e-8)).
// Each lane's 16 values are two runs of 8 k's of one weight row, 32B write.
// mat 0 = Ww1 (w_aff_W cols [0,HALF))      -> white contribution to w_
// mat 1 = Ww2 (w_aff_W cols [HALF,NUMEL))  -> black contribution to w_
// mat 2 = Wb1 (b_aff_W cols [0,HALF))      -> black contribution to b_
// mat 3 = Wb2 (b_aff_W cols [HALF,NUMEL))  -> white contribution to b_
// ---------------------------------------------------------------------------
__global__ __launch_bounds__(256) void nnue_convert_weights(
    const float* __restrict__ wW, const float* __restrict__ bW,
    __bf16* __restrict__ frag) {
  const int f = blockIdx.x * blockDim.x + threadIdx.x;   // 0..786431
  const int m    = f / 196608;                           // matrix 0..3
  const int fm   = f % 196608;                           // KTILES*NTILES*32
  const int kt   = fm >> 8;
  const int rem  = fm & 255;
  const int nt   = rem >> 5;
  const int lane = rem & 31;
  const int n  = nt * 16 + (lane & 15);
  const int g  = lane >> 4;
  const int k0 = kt * 32 + g * 8;
  const float* src = (m < 2) ? wW : bW;
  const int col0 = (m & 1) ? (HALFK + k0) : k0;
  const float* p = src + (size_t)n * NUMELK + col0;
  v16bf v;
#pragma unroll
  for (int e = 0; e < 16; ++e) v[e] = (__bf16)p[e + (e >= 8 ? 8 : 0)];
  *(v16bf*)(frag + (size_t)f * 16) = v;
}

// ---------------------------------------------------------------------------
// async copy helpers (gfx1250): 16B global -> LDS, ASYNCcnt-tracked.
// ---------------------------------------------------------------------------
#define ASYNC_CP16(ldsoff, gptr, IMM)                                      \
  asm volatile("global_load_async_to_lds_b128 %0, %1, off offset:" IMM     \
               :: "v"(ldsoff), "v"(gptr) : "memory")

__device__ __forceinline__ void wait_async_le8() {
  asm volatile("s_wait_asynccnt 0x8" ::: "memory");
}
__device__ __forceinline__ void wait_async_0() {
  asm volatile("s_wait_asynccnt 0x0" ::: "memory");
}

// Stage s (0..767) covers k-steps ktt = 2s, 2s+1 (side = s>=384).
// Copies 2 contiguous 16KB regions (w_-weights, b_-weights) into LDS.
// 256 threads: t<128 -> w_ region, t>=128 -> b_ region; 128B (8 x B128) each.
__device__ __forceinline__ void issue_stage(const char* fragBase, int s,
                                            unsigned ldsStage, int tid) {
  const int side = (s >= 384) ? 1 : 0;
  const size_t ktoff = (size_t)(2 * s - side * KTILES) * 8192;
  const char* gW = fragBase + (size_t)(side ? 1 : 0) * FRAG_BYTES + ktoff;
  const char* gB = fragBase + (size_t)(side ? 2 : 3) * FRAG_BYTES + ktoff;
  const int t = tid & 127;
  const char* g = ((tid < 128) ? gW : gB) + (size_t)t * 128;
  const unsigned lds = ldsStage + ((tid < 128) ? 0u : 16384u) + (unsigned)t * 128u;
  ASYNC_CP16(lds, g, "0");
  ASYNC_CP16(lds, g, "16");
  ASYNC_CP16(lds, g, "32");
  ASYNC_CP16(lds, g, "48");
  ASYNC_CP16(lds, g, "64");
  ASYNC_CP16(lds, g, "80");
  ASYNC_CP16(lds, g, "96");
  ASYNC_CP16(lds, g, "112");
}

// ---------------------------------------------------------------------------
// Kernel 2: the big GEMM. 64 WGs x 256 threads (8 waves).
// Wave w owns rows [blk*128 + w*16, +16) and all 128 output columns for BOTH
// w_ and b_ (16 f32 accumulator tiles = 128 VGPRs). One feature load feeds
// two WMMAs. Weight fragments are double-buffered in LDS (2 stages x 32KB),
// filled by async-to-LDS copies shared across the 8 waves.
// A-matrix 16x32 bf16 ISA layout: lane L, mrow = L&15, h = L>>4:
//   e<8  -> k = kt*32 + h*8 + e ; e>=8 -> k = kt*32 + 16 + h*8 + (e-8)
// ---------------------------------------------------------------------------
__global__ __launch_bounds__(256) void nnue_gemm(
    const float* __restrict__ white, const float* __restrict__ black,
    const __bf16* __restrict__ frag,
    const float* __restrict__ wBias, const float* __restrict__ bBias,
    float* __restrict__ wOut, float* __restrict__ bOut) {
  // [stage][mat][2 k-steps * 8 nt * 32 lanes * 16 bf16]
  __shared__ __attribute__((aligned(16))) __bf16 sTiles[2][2][8192];

  const int tid   = threadIdx.x;
  const int lane  = tid & 31;
  const int wid   = tid >> 5;
  const int mBase = blockIdx.x * 128 + wid * 16;
  const int row   = mBase + (lane & 15);
  const int h     = lane >> 4;

  const float* rowW = white + (size_t)row * HALFK + h * 8;
  const float* rowB = black + (size_t)row * HALFK + h * 8;

  v8f accW[NTILES], accB[NTILES];
#pragma unroll
  for (int i = 0; i < NTILES; ++i) {
    accW[i] = (v8f){0.f,0.f,0.f,0.f,0.f,0.f,0.f,0.f};
    accB[i] = (v8f){0.f,0.f,0.f,0.f,0.f,0.f,0.f,0.f};
  }

  const unsigned ldsBase = (unsigned)(uintptr_t)&sTiles[0][0][0];
  const char* fragBase = (const char*)frag;

  issue_stage(fragBase, 0, ldsBase, tid);               // prefetch stage 0

  for (int ss = 0; ss < NSTAGES; ++ss) {
    if (ss + 1 < NSTAGES) {
      issue_stage(fragBase, ss + 1, ldsBase + ((unsigned)((ss + 1) & 1)) * 32768u, tid);
      wait_async_le8();     // current stage's 8 copies done; next 8 in flight
    } else {
      wait_async_0();
    }
    __syncthreads();        // all waves' copies for this stage resident

    const int st = ss & 1;
#pragma unroll
    for (int k2 = 0; k2 < 2; ++k2) {
      const int ktt  = ss * 2 + k2;
      const int side = (ktt >= KTILES) ? 1 : 0;         // uniform
      const int kt   = ktt - side * KTILES;
      const float* a0 = (side ? rowB : rowW) + kt * 32;
      __builtin_prefetch(a0 + 512, 0, 1);               // 16 k-steps ahead

      v4f f0 = *(const v4f*)(a0);
      v4f f1 = *(const v4f*)(a0 + 4);
      v4f f2 = *(const v4f*)(a0 + 16);
      v4f f3 = *(const v4f*)(a0 + 20);
      v16bf a;
#pragma unroll
      for (int e = 0; e < 4; ++e) {
        a[e]      = (__bf16)f0[e];
        a[e + 4]  = (__bf16)f1[e];
        a[e + 8]  = (__bf16)f2[e];
        a[e + 12] = (__bf16)f3[e];
      }

      const v16bf* pW = (const v16bf*)&sTiles[st][0][k2 * 4096];
      const v16bf* pB = (const v16bf*)&sTiles[st][1][k2 * 4096];
#pragma unroll
      for (int nt = 0; nt < NTILES; ++nt) {
        v16bf bw = pW[nt * 32 + lane];
        accW[nt] = __builtin_amdgcn_wmma_f32_16x16x32_bf16(
            false, a, false, bw, (short)0, accW[nt], false, false);
        v16bf bb = pB[nt * 32 + lane];
        accB[nt] = __builtin_amdgcn_wmma_f32_16x16x32_bf16(
            false, a, false, bb, (short)0, accB[nt], false, false);
      }
      // Pipeline choreography for this k-step:
      //   DS_READx4 (tile-pair 0), then 7x { WMMAx2 || DS_READx4 (pair i+1) },
      //   then the final WMMAx2. Keeps two tile-pairs in flight so the
      //   ds_load -> wmma dependency never drains to s_wait_dscnt 0.
      SGB(0x100, 4, 0);                 // 4 ds reads (tile pair 0)
#pragma unroll
      for (int i = 0; i < NTILES - 1; ++i) {
        SGB(0x8, 2, 0);                 // 2 WMMAs (tile pair i)
        SGB(0x100, 4, 0);               // 4 ds reads (tile pair i+1)
      }
      SGB(0x8, 2, 0);                   // final 2 WMMAs
    }
    __syncthreads();        // all waves done reading before next overwrite
  }

  // Epilogue: D tile layout -> VGPR r, lanes 0-15: (m = mBase + r, n = lane),
  // lanes 16-31: (m = mBase + 8 + r, n = lane-16). Add bias, store f32.
#pragma unroll
  for (int nt = 0; nt < NTILES; ++nt) {
    const int n = nt * 16 + (lane & 15);
    const float bw = wBias[n];
    const float bb = bBias[n];
#pragma unroll
    for (int r = 0; r < 8; ++r) {
      const int m = mBase + h * 8 + r;
      wOut[(size_t)m * BASEN + n] = accW[nt][r] + bw;
      bOut[(size_t)m * BASEN + n] = accB[nt][r] + bb;
    }
  }
}

// ---------------------------------------------------------------------------
// Kernel 3: POV mix + ReLU + fc0..fc3 (tiny). One wave per batch row; fc
// weights staged in LDS once per block. 256 blocks x 8 waves -> every wave
// does exactly 4 rows => uniform __syncthreads.
// ---------------------------------------------------------------------------
__global__ __launch_bounds__(256) void nnue_tail(
    const float* __restrict__ pov,
    const float* __restrict__ wbuf, const float* __restrict__ bbuf,
    const float* __restrict__ fc0W, const float* __restrict__ fc0b,
    const float* __restrict__ fc1W, const float* __restrict__ fc1b,
    const float* __restrict__ fc2W, const float* __restrict__ fc2b,
    const float* __restrict__ fc3W, const float* __restrict__ fc3b,
    float* __restrict__ out) {
  __shared__ float sW0[32 * 256];
  __shared__ float sW1[32 * 32];
  __shared__ float sW2[32 * 64];
  __shared__ float sW3[96];
  __shared__ float sB0[32], sB1[32], sB2[32], sB3[1];
  __shared__ float sBase[8][256];
  __shared__ float sX[8][96];

  const int tid = threadIdx.x;
  for (int i = tid; i < 32 * 256; i += 256) sW0[i] = fc0W[i];
  for (int i = tid; i < 32 * 32;  i += 256) sW1[i] = fc1W[i];
  for (int i = tid; i < 32 * 64;  i += 256) sW2[i] = fc2W[i];
  if (tid < 96) sW3[tid] = fc3W[tid];
  if (tid < 32) { sB0[tid] = fc0b[tid]; sB1[tid] = fc1b[tid]; sB2[tid] = fc2b[tid]; }
  if (tid == 0) sB3[0] = fc3b[0];
  __syncthreads();

  const int lane = tid & 31;
  const int wid  = tid >> 5;

  for (int row = blockIdx.x * 8 + wid; row < BATCH; row += gridDim.x * 8) {
    const float p = pov[row];
    const float* wr = wbuf + (size_t)row * BASEN;
    const float* br = bbuf + (size_t)row * BASEN;
    for (int j = lane; j < 256; j += 32) {
      const float wbv = (j < 128) ? wr[j] : br[j - 128];
      const float bwv = (j < 128) ? br[j] : wr[j - 128];
      const float v = p * wbv + (1.0f - p) * bwv;
      sBase[wid][j] = v > 0.f ? v : 0.f;
    }
    __syncthreads();
    {   // fc0: 32 outputs over 256 inputs
      float acc = sB0[lane];
      const float* wn = &sW0[lane * 256];
      for (int j = 0; j < 256; ++j) acc += wn[j] * sBase[wid][j];
      sX[wid][lane] = acc > 0.f ? acc : 0.f;
    }
    __syncthreads();
    {   // fc1: x = cat(x0, relu(fc1(x0)))
      float acc = sB1[lane];
      const float* wn = &sW1[lane * 32];
      for (int j = 0; j < 32; ++j) acc += wn[j] * sX[wid][j];
      sX[wid][32 + lane] = acc > 0.f ? acc : 0.f;
    }
    __syncthreads();
    {   // fc2: x = cat(x, relu(fc2(x)))
      float acc = sB2[lane];
      const float* wn = &sW2[lane * 64];
      for (int j = 0; j < 64; ++j) acc += wn[j] * sX[wid][j];
      sX[wid][64 + lane] = acc > 0.f ? acc : 0.f;
    }
    __syncthreads();
    if (lane == 0) {  // fc3: scalar output
      float acc = sB3[0];
      for (int j = 0; j < 96; ++j) acc += sW3[j] * sX[wid][j];
      out[row] = acc;
    }
    __syncthreads();
  }
}

// ---------------------------------------------------------------------------
// Launch. Workspace layout (32 MB total):
//   [0, 25165824)          : 4 bf16 weight-fragment matrices (6.29 MB each)
//   [25165824, 29360128)   : w_ intermediate, f32 [8192 x 128]
//   [29360128, 33554432)   : b_ intermediate, f32 [8192 x 128]
// ---------------------------------------------------------------------------
extern "C" void kernel_launch(void* const* d_in, const int* in_sizes, int n_in,
                              void* d_out, int out_size, void* d_ws, size_t ws_size,
                              hipStream_t stream) {
  const float* pov     = (const float*)d_in[0];
  const float* white   = (const float*)d_in[1];
  const float* black   = (const float*)d_in[2];
  const float* w_aff_W = (const float*)d_in[3];
  const float* w_aff_b = (const float*)d_in[4];
  const float* b_aff_W = (const float*)d_in[5];
  const float* b_aff_b = (const float*)d_in[6];
  const float* fc0W    = (const float*)d_in[7];
  const float* fc0b    = (const float*)d_in[8];
  const float* fc1W    = (const float*)d_in[9];
  const float* fc1b    = (const float*)d_in[10];
  const float* fc2W    = (const float*)d_in[11];
  const float* fc2b    = (const float*)d_in[12];
  const float* fc3W    = (const float*)d_in[13];
  const float* fc3b    = (const float*)d_in[14];
  float* out = (float*)d_out;

  char* ws = (char*)d_ws;
  __bf16* frag = (__bf16*)ws;
  float* wbuf = (float*)(ws + 4 * FRAG_BYTES);
  float* bbuf = (float*)(ws + 4 * FRAG_BYTES + (size_t)BATCH * BASEN * 4);

  // 1) repack weights to bf16 fragments: 786432 fragments, one per thread
  nnue_convert_weights<<<3072, 256, 0, stream>>>(w_aff_W, b_aff_W, frag);

  // 2) the big fused dual-output GEMM: 8192 rows / 128 per WG
  nnue_gemm<<<BATCH / 128, 256, 0, stream>>>(white, black, frag,
                                             w_aff_b, b_aff_b, wbuf, bbuf);

  // 3) tail MLP
  nnue_tail<<<256, 256, 0, stream>>>(pov, wbuf, bbuf, fc0W, fc0b, fc1W, fc1b,
                                     fc2W, fc2b, fc3W, fc3b, out);
}
